// Net_43344809952018
// MI455X (gfx1250) — compile-verified, matrix operand order checked by reading, not run on hardware
//
#include <hip/hip_runtime.h>

// Problem constants
#define BG 8
#define NN 4096
#define KNB 24
#define HD 32
#define NNODE (BG * NN)          // 32768
#define SSTRIDE 4100             // padded LDS score row stride (dwords): no bank conflicts
#define ASTRIDE 34               // padded A-stage stride (dwords)
#define BSTRIDE 36               // padded async B-tile stride (dwords, 16B-aligned rows)
#define KNN_LDS (16 * SSTRIDE * 4 + 16 * ASTRIDE * 4 + 8 * 2 * 16 * BSTRIDE * 4)  // 301440 B

typedef __attribute__((ext_vector_type(2)))  float    v2f;
typedef __attribute__((ext_vector_type(8)))  float    v8f;
typedef __attribute__((ext_vector_type(16))) _Float16 v16h;
typedef int v4i_ __attribute__((vector_size(16)));   // matches builtin param type

// ---- CDNA5 async global->LDS copy (ASYNCcnt-tracked), builtin or asm ----
#if __has_builtin(__builtin_amdgcn_global_load_async_to_lds_b128)
#define ASYNC_CP128(g, l)                                                              \
  __builtin_amdgcn_global_load_async_to_lds_b128(                                      \
      (__attribute__((address_space(1))) v4i_*)(unsigned long long)(const void*)(g),   \
      (__attribute__((address_space(3))) v4i_*)(unsigned)(unsigned long long)(const void*)(l), \
      0, 0)
#else
#define ASYNC_CP128(g, l)                                                              \
  asm volatile("global_load_async_to_lds_b128 %0, %1, off"                             \
               :: "v"((unsigned)(unsigned long long)(const void*)(l)),                 \
                  "v"((unsigned long long)(const void*)(g))                            \
               : "memory")
#endif

#if __has_builtin(__builtin_amdgcn_s_wait_asynccnt)
#define ASYNC_WAIT(n) __builtin_amdgcn_s_wait_asynccnt(n)
#else
#define ASYNC_WAIT(n) asm volatile("s_wait_asynccnt %0" :: "i"(n) : "memory")
#endif

__device__ __forceinline__ float eluf(float x) { return x > 0.0f ? x : (__expf(x) - 1.0f); }

// ------------------------------------------------------------------
// Encoder: [32768,4] -> elu -> [32768,32] -> elu -> [32768,32]
// ------------------------------------------------------------------
__global__ __launch_bounds__(256) void encode_kernel(
    const float* __restrict__ x, const float* __restrict__ w1, const float* __restrict__ b1,
    const float* __restrict__ w2, const float* __restrict__ b2, float* __restrict__ Y)
{
  int i = blockIdx.x * 256 + threadIdx.x;
  float in0 = x[i * 4 + 0], in1 = x[i * 4 + 1], in2 = x[i * 4 + 2], in3 = x[i * 4 + 3];
  float h1[HD];
  #pragma unroll
  for (int o = 0; o < HD; ++o)
    h1[o] = eluf(b1[o] + in0 * w1[0 * HD + o] + in1 * w1[1 * HD + o] +
                 in2 * w1[2 * HD + o] + in3 * w1[3 * HD + o]);
  for (int o = 0; o < HD; ++o) {
    float a = b2[o];
    #pragma unroll
    for (int c = 0; c < HD; ++c) a += h1[c] * w2[c * HD + o];
    Y[(size_t)i * HD + o] = eluf(a);
  }
}

// ------------------------------------------------------------------
// Squared norms per node
// ------------------------------------------------------------------
__global__ __launch_bounds__(256) void d2_kernel(const float* __restrict__ X, float* __restrict__ d2)
{
  int i = blockIdx.x * 256 + threadIdx.x;
  const float* p = X + (size_t)i * HD;
  float s = 0.0f;
  #pragma unroll
  for (int c = 0; c < HD; ++c) s += p[c] * p[c];
  d2[i] = s;
}

// ------------------------------------------------------------------
// kNN kernel: one workgroup = one graph x 16-row tile.
// GEMM phase: B column tiles (contiguous 2KB slabs of X) are double-
// buffered into LDS with async global->LDS b128 copies (ASYNCcnt),
// then f32 WMMA 16x16x4 (two independent 4-deep chains over K=32).
// Score s = d2[j] - 2*dot goes to a 16x4096 LDS strip (258KB — only
// possible with CDNA5's 320KB WGP LDS).
// Select phase: 2 rows per wave, 24 wave-argmin passes (shfl_xor).
// ------------------------------------------------------------------
__global__ __launch_bounds__(256) void knn_kernel(
    const float* __restrict__ X, const float* __restrict__ d2g, int* __restrict__ idxout)
{
  extern __shared__ char smem[];
  float* scores = (float*)smem;                                        // [16][SSTRIDE]
  float* Asm    = (float*)(smem + 16 * SSTRIDE * 4);                   // [16][ASTRIDE]
  float* Bbase  = (float*)(smem + 16 * SSTRIDE * 4 + 16 * ASTRIDE * 4);// [8][2][16*BSTRIDE]

  const int b = blockIdx.y;
  const int rowbase = blockIdx.x * 16;
  const float* Xb  = X + (size_t)b * NN * HD;
  const float* d2b = d2g + (size_t)b * NN;
  const int tid = threadIdx.x;
  const int lane = tid & 31;
  const int wave = tid >> 5;

  float* bufs = Bbase + (size_t)wave * 2 * 16 * BSTRIDE;  // per-wave double buffer

  // Stage the 16 A rows (16x32 f32) into LDS
  for (int t = tid; t < 16 * HD; t += 256) {
    int r = t >> 5, k = t & 31;
    Asm[r * ASTRIDE + k] = Xb[(size_t)(rowbase + r) * HD + k];
  }
  __syncthreads();

  const int r16  = lane & 15;
  const int koff = (lane >> 4) * 2;   // f32 A/B fragment K sub-offset

  // Preload 8 A fragments (K = 0..31 in steps of 4)
  v2f afrag[8];
  #pragma unroll
  for (int s = 0; s < 8; ++s) {
    afrag[s].x = Asm[r16 * ASTRIDE + s * 4 + koff];
    afrag[s].y = Asm[r16 * ASTRIDE + s * 4 + koff + 1];
  }

  // Async-stage one 16-column tile (2KB contiguous): lane copies 4 x 16B chunks,
  // scattered into a bank-conflict-free padded layout.
  auto stage_tile = [&](int colbase, float* dst) {
    const char* gsrc = (const char*)(Xb + (size_t)colbase * HD);
    #pragma unroll
    for (int q = 0; q < 4; ++q) {
      int c = lane + 32 * q;            // chunk id 0..127
      int row = c >> 3, col = c & 7;    // 8 chunks per 128B row
      ASYNC_CP128(gsrc + (size_t)c * 16, dst + row * BSTRIDE + col * 4);
    }
  };

  // Column tiles: wave w owns tiles w, w+8, ..., w+248 (32 tiles), pipelined.
  stage_tile(wave * 16, bufs);
  for (int i = 0; i < 32; ++i) {
    const int t = wave + i * 8;
    const int colbase = t * 16;
    float* cur = bufs + (i & 1) * 16 * BSTRIDE;
    if (i + 1 < 32) {
      stage_tile((t + 8) * 16, bufs + ((i + 1) & 1) * 16 * BSTRIDE);
      ASYNC_WAIT(4);   // 4 newest (prefetch) may remain in flight; cur is complete
    } else {
      ASYNC_WAIT(0);
    }

    const float* xcol = cur + r16 * BSTRIDE;
    v8f acc0 = {}, acc1 = {};
    #pragma unroll
    for (int s = 0; s < 4; ++s) {  // two independent WMMA chains for ILP
      v2f b0, b1;
      b0.x = xcol[(2 * s) * 4 + koff];
      b0.y = xcol[(2 * s) * 4 + koff + 1];
      b1.x = xcol[(2 * s + 1) * 4 + koff];
      b1.y = xcol[(2 * s + 1) * 4 + koff + 1];
      acc0 = __builtin_amdgcn_wmma_f32_16x16x4_f32(
          false, afrag[2 * s], false, b0, (short)0, acc0, false, false);
      acc1 = __builtin_amdgcn_wmma_f32_16x16x4_f32(
          false, afrag[2 * s + 1], false, b1, (short)0, acc1, false, false);
    }
    const float d2n = d2b[colbase + r16];
    const int rbase = (lane >> 4) * 8;
    #pragma unroll
    for (int r8 = 0; r8 < 8; ++r8) {
      float s = d2n - 2.0f * (acc0[r8] + acc1[r8]);
      scores[(r8 + rbase) * SSTRIDE + colbase + r16] = s;
    }
  }
  __syncthreads();

  // Top-K selection: wave w owns rows 2w, 2w+1
  for (int rr = 0; rr < 2; ++rr) {
    const int row = wave * 2 + rr;
    float* srow = scores + row * SSTRIDE;
    for (int k = 0; k < KNB; ++k) {
      float best = 3.0e38f;
      int bestj = 0x7fffffff;
      for (int j = lane; j < NN; j += 32) {
        float v = srow[j];
        if (v < best) { best = v; bestj = j; }   // strict <, keeps lowest index
      }
      #pragma unroll
      for (int off = 16; off > 0; off >>= 1) {
        float ov = __shfl_xor(best, off, 32);
        int   oj = __shfl_xor(bestj, off, 32);
        if (ov < best || (ov == best && oj < bestj)) { best = ov; bestj = oj; }
      }
      if (lane == 0)
        idxout[((size_t)b * NN + rowbase + row) * KNB + k] = bestj;
      if ((bestj & 31) == lane) srow[bestj] = 3.0e38f;  // knock out winner (in-order DS, same wave)
    }
  }
}

// ------------------------------------------------------------------
// EdgeConv kernel: one workgroup = 32 nodes = 768 edges = 48 tiles of 16.
// Gather edge features [x_i | x_j - x_i] (f16) into per-wave LDS stage,
// f16 WMMA 16x16x32 (two independent K-steps) x2 N-tiles per tile,
// ELU + bias, max-aggregate via monotone-uint ds_max atomics in LDS.
// ------------------------------------------------------------------
__global__ __launch_bounds__(256) void edge_kernel(
    const float* __restrict__ X, const int* __restrict__ idx,
    const float* __restrict__ w, const float* __restrict__ bias, float* __restrict__ Y)
{
  __shared__ _Float16 Wt[32][64];          // weight, [n][k] (pre-transposed)
  __shared__ float    bsh[32];
  __shared__ unsigned nodeacc[32][32];     // encoded-float max accumulators
  __shared__ _Float16 Astage[8][16 * 64];  // per-wave edge-feature stage

  const int tid  = threadIdx.x;
  const int lane = tid & 31;
  const int wave = tid >> 5;
  const int nodebase = blockIdx.x * 32;

  // Stage W^T (w is [64][32] row-major, k-major) and bias; init accumulators
  for (int t = tid; t < 64 * 32; t += 256) {
    int n = t & 31, k = t >> 5;
    Wt[n][k] = (_Float16)w[k * 32 + n];
  }
  if (tid < 32) bsh[tid] = bias[tid];
  for (int t = tid; t < 32 * 32; t += 256) nodeacc[t >> 5][t & 31] = 0u;
  __syncthreads();

  const int l15 = lane & 15;
  const int kb  = (lane >> 4) * 8;   // f16 fragment K sub-offset

  // B fragments: 2 N-tiles x 2 K-steps, invariant across edge tiles
  v16h bfrag[2][2];
  #pragma unroll
  for (int nt = 0; nt < 2; ++nt) {
    #pragma unroll
    for (int s = 0; s < 2; ++s) {
      const int n = nt * 16 + l15;
      #pragma unroll
      for (int e = 0; e < 8; ++e) {
        bfrag[nt][s][e]     = Wt[n][s * 32 + kb + e];
        bfrag[nt][s][8 + e] = Wt[n][s * 32 + 16 + kb + e];
      }
    }
  }

  _Float16* A = Astage[wave];

  for (int t = wave; t < 48; t += 8) {
    const int e0 = t * 16;  // block-local edge id base
    // Gather: lane handles half a row (32 features)
    {
      const int row  = lane >> 1;
      const int half = lane & 1;
      const int e    = e0 + row;
      const int nl   = e / KNB;
      const int kk   = e - nl * KNB;
      const int inode = nodebase + nl;
      const int bg    = inode >> 12;
      const int jg    = idx[(size_t)inode * KNB + kk];
      const int jnode = (bg << 12) + jg;
      const float* xi = X + (size_t)inode * HD;
      const float* xj = X + (size_t)jnode * HD;
      if (half == 0) {
        #pragma unroll
        for (int c = 0; c < 32; ++c) A[row * 64 + c] = (_Float16)xi[c];
      } else {
        #pragma unroll
        for (int c = 0; c < 32; ++c) A[row * 64 + 32 + c] = (_Float16)(xj[c] - xi[c]);
      }
    }
    asm volatile("s_wait_dscnt 0" ::: "memory");  // wave-private stage: DS in-order per wave

    // A fragments: 2 K-steps of 16x32
    v16h afrag[2];
    #pragma unroll
    for (int s = 0; s < 2; ++s) {
      #pragma unroll
      for (int e = 0; e < 8; ++e) {
        afrag[s][e]     = A[l15 * 64 + s * 32 + kb + e];
        afrag[s][8 + e] = A[l15 * 64 + s * 32 + 16 + kb + e];
      }
    }

    #pragma unroll
    for (int nt = 0; nt < 2; ++nt) {
      v8f c0 = {}, c1 = {};  // independent chains
      c0 = __builtin_amdgcn_wmma_f32_16x16x32_f16(false, afrag[0], false, bfrag[nt][0],
                                                  (short)0, c0, false, false);
      c1 = __builtin_amdgcn_wmma_f32_16x16x32_f16(false, afrag[1], false, bfrag[nt][1],
                                                  (short)0, c1, false, false);
      const int n = nt * 16 + l15;
      const float bv = bsh[n];
      const int rbase = (lane >> 4) * 8;
      #pragma unroll
      for (int r8 = 0; r8 < 8; ++r8) {
        const int e  = e0 + r8 + rbase;
        const int nl = e / KNB;
        float v = eluf(c0[r8] + c1[r8] + bv);
        unsigned u = __float_as_uint(v);
        u = (u & 0x80000000u) ? ~u : (u | 0x80000000u);  // monotone encode
        atomicMax(&nodeacc[nl][n], u);                   // ds_max_u32
      }
    }
  }
  __syncthreads();

  // Decode and write node features
  for (int t = tid; t < 32 * 32; t += 256) {
    const int nl = t >> 5, n = t & 31;
    unsigned u = nodeacc[nl][n];
    unsigned bits = (u & 0x80000000u) ? (u ^ 0x80000000u) : ~u;
    Y[(size_t)(nodebase + nl) * HD + n] = __uint_as_float(bits);
  }
}

// ------------------------------------------------------------------
// Output MLP 32 -> 32 -> 16 -> 8, plus second tuple output (batch ids)
// ------------------------------------------------------------------
__global__ __launch_bounds__(256) void out_kernel(
    const float* __restrict__ Xf,
    const float* __restrict__ w1, const float* __restrict__ b1,
    const float* __restrict__ w2, const float* __restrict__ b2,
    const float* __restrict__ w3, const float* __restrict__ b3,
    const long long* __restrict__ batch, float* __restrict__ out, int out_size)
{
  int i = blockIdx.x * 256 + threadIdx.x;
  float h[HD];
  #pragma unroll
  for (int c = 0; c < HD; ++c) h[c] = Xf[(size_t)i * HD + c];
  float o1[32];
  for (int o = 0; o < 32; ++o) {
    float a = b1[o];
    #pragma unroll
    for (int c = 0; c < HD; ++c) a += h[c] * w1[c * 32 + o];
    o1[o] = eluf(a);
  }
  float o2[16];
  for (int o = 0; o < 16; ++o) {
    float a = b2[o];
    #pragma unroll
    for (int c = 0; c < 32; ++c) a += o1[c] * w2[c * 16 + o];
    o2[o] = eluf(a);
  }
  #pragma unroll
  for (int o = 0; o < 8; ++o) {
    float a = b3[o];
    #pragma unroll
    for (int c = 0; c < 16; ++c) a += o2[c] * w3[c * 8 + o];
    out[(size_t)i * 8 + o] = a;
  }
  const int OFLT = NNODE * 8;
  if (out_size >= OFLT + 2 * NNODE) {
    long long* tail = (long long*)(out + OFLT);
    tail[i] = batch[i];
  } else if (out_size >= OFLT + NNODE) {
    out[OFLT + i] = (float)batch[i];
  }
}

// ------------------------------------------------------------------
extern "C" void kernel_launch(void* const* d_in, const int* in_sizes, int n_in,
                              void* d_out, int out_size, void* d_ws, size_t ws_size,
                              hipStream_t stream)
{
  const float*     x_lc     = (const float*)d_in[0];
  const long long* batch_lc = (const long long*)d_in[1];
  const float* enc_w1 = (const float*)d_in[2];
  const float* enc_b1 = (const float*)d_in[3];
  const float* enc_w2 = (const float*)d_in[4];
  const float* enc_b2 = (const float*)d_in[5];
  const float* conv_w[3] = { (const float*)d_in[6], (const float*)d_in[8],  (const float*)d_in[10] };
  const float* conv_b[3] = { (const float*)d_in[7], (const float*)d_in[9],  (const float*)d_in[11] };
  const float* out_w1 = (const float*)d_in[12];
  const float* out_b1 = (const float*)d_in[13];
  const float* out_w2 = (const float*)d_in[14];
  const float* out_b2 = (const float*)d_in[15];
  const float* out_w3 = (const float*)d_in[16];
  const float* out_b3 = (const float*)d_in[17];

  // Workspace layout
  float* feat0 = (float*)d_ws;                 // 32768*32
  float* feat1 = feat0 + (size_t)NNODE * HD;   // 32768*32
  float* d2buf = feat1 + (size_t)NNODE * HD;   // 32768
  int*   idxb  = (int*)(d2buf + NNODE);        // 32768*24

  // Allow >64KB dynamic LDS for the kNN kernel (needs CDNA5's 320KB WGP LDS)
  (void)hipFuncSetAttribute(reinterpret_cast<const void*>(knn_kernel),
                            hipFuncAttributeMaxDynamicSharedMemorySize, KNN_LDS);

  float* out = (float*)d_out;

  encode_kernel<<<NNODE / 256, 256, 0, stream>>>(x_lc, enc_w1, enc_b1, enc_w2, enc_b2, feat0);

  float* cur = feat0;
  float* nxt = feat1;
  for (int layer = 0; layer < 3; ++layer) {
    d2_kernel<<<NNODE / 256, 256, 0, stream>>>(cur, d2buf);
    knn_kernel<<<dim3(NN / 16, BG), 256, KNN_LDS, stream>>>(cur, d2buf, idxb);
    edge_kernel<<<NNODE / 32, 256, 0, stream>>>(cur, idxb, conv_w[layer], conv_b[layer], nxt);
    float* t = cur; cur = nxt; nxt = t;
  }

  out_kernel<<<NNODE / 256, 256, 0, stream>>>(cur, out_w1, out_b1, out_w2, out_b2,
                                              out_w3, out_b3, batch_lc, out, out_size);
}
